// GraphormerLayer_54863912239506
// MI455X (gfx1250) — matrix operand order repaired and checked by reading, not compiled, not run on hardware
//
#include <hip/hip_runtime.h>

typedef __attribute__((ext_vector_type(16))) _Float16 v16h;
typedef __attribute__((ext_vector_type(8)))  _Float16 v8h;
typedef __attribute__((ext_vector_type(8)))  float    v8f;

#define DEV static __device__ __forceinline__

// Try the gfx1250 async global->LDS copy path (GLOBAL_LOAD_ASYNC_TO_LDS_B128).
#define USE_ASYNC_COPY 1

constexpr int B_  = 8;
constexpr int N_  = 512;
constexpr int E_  = 512;
constexpr int H_  = 8;
constexpr int NE_ = 16384;
constexpr int DH_ = 64;
constexpr int MROWS = B_ * N_;          // 4096
constexpr float SCALE_ = 0.125f;        // DH^-0.5
constexpr int BPITCH = 40;              // LDS pitch (halves) for 32-wide K tile

// ---------------------------------------------------------------------------
// WMMA fragment loaders (wave32, 16x16x32 f16 -> f32)
// A (16xK tile, row-major, arbitrary row stride):
//   lane: m = lane&15, hs = lane>>4
//   elem i in [0,16): k = k0 + hs*16 + i (contiguous)
DEV v16h load_a_frag(const _Float16* rowBase, int stride, int k0, int lane) {
  int m = lane & 15, hs = lane >> 4;
  const _Float16* p = rowBase + (size_t)m * stride + k0 + hs * 16;
  v8h lo = *(const v8h*)p;
  v8h hi = *(const v8h*)(p + 8);
  v16h a;
#pragma unroll
  for (int i = 0; i < 8; ++i) { a[i] = lo[i]; a[i + 8] = hi[i]; }
  return a;
}

// B (Kx16 tile where B[k][n] = W[n][k], W row-major [Nout,K]):
//   lane: n = lane&15, hs = lane>>4 ; elem i: k = k0 + hs*16 + i (contiguous)
DEV v16h load_b_frag(const _Float16* colBase, int stride, int k0, int lane) {
  int n = lane & 15, hs = lane >> 4;
  return *(const v16h*)(colBase + (size_t)n * stride + k0 + hs * 16);
}

// B fragment from an LDS-staged 64xBPITCH tile (K-step fixed at 32)
DEV v16h load_b_lds(const _Float16* base, int ncol0, int lane) {
  int n = lane & 15, hs = lane >> 4;
  return *(const v16h*)(base + (size_t)(ncol0 + n) * BPITCH + hs * 16);
}

DEV v8f wmma_f16(v16h a, v16h b, v8f c) {
  return __builtin_amdgcn_wmma_f32_16x16x32_f16(false, a, false, b, (short)0, c,
                                                false, false);
}

// ---------------------------------------------------------------------------
// f32 -> f16 cast
__global__ __launch_bounds__(256) void cast_f16_kernel(const float* __restrict__ in,
                                                       _Float16* __restrict__ out,
                                                       int n) {
  int i = blockIdx.x * 256 + threadIdx.x;
  if (i < n) out[i] = (_Float16)in[i];
}

// ---------------------------------------------------------------------------
// LayerNorm over E=512, one block (256 thr) per row, f16 output
__global__ __launch_bounds__(256) void layernorm_kernel(const float* __restrict__ x,
                                                        const float* __restrict__ g,
                                                        const float* __restrict__ be,
                                                        _Float16* __restrict__ out) {
  int row = blockIdx.x;
  int t = threadIdx.x;
  const float* xr = x + (size_t)row * E_;
  float a0 = xr[t], a1 = xr[t + 256];
  float s = a0 + a1, s2 = a0 * a0 + a1 * a1;
#pragma unroll
  for (int o = 16; o > 0; o >>= 1) {
    s  += __shfl_down(s,  o, 32);
    s2 += __shfl_down(s2, o, 32);
  }
  __shared__ float rs[8], rs2[8];
  int lane = t & 31, wv = t >> 5;
  if (lane == 0) { rs[wv] = s; rs2[wv] = s2; }
  __syncthreads();
  float S = 0.f, S2 = 0.f;
#pragma unroll
  for (int i = 0; i < 8; ++i) { S += rs[i]; S2 += rs2[i]; }
  float mu  = S * (1.f / E_);
  float var = S2 * (1.f / E_) - mu * mu;
  float rstd = rsqrtf(var + 1e-5f);
  _Float16* o = out + (size_t)row * E_;
  o[t]       = (_Float16)((a0 - mu) * rstd * g[t] + be[t]);
  o[t + 256] = (_Float16)((a1 - mu) * rstd * g[t + 256] + be[t + 256]);
}

// ---------------------------------------------------------------------------
// WMMA GEMM: out[M,N] = A[M,K] @ W[N,K]^T + bias, with epilogue.
// EPI 0: f16 out. EPI 1: GELU(exact) -> f16 out. EPI 2: + resid -> f32 out.
// Block: 256 thr = 8 waves -> 256x64 tile; each wave: 32x64 (2x4 accums).
// Weight tile (64 cols x 32 k) staged in LDS, double-buffered; staging uses
// gfx1250 async global->LDS copies so the fetch of tile t+1 overlaps the
// WMMAs on tile t.
template <int EPI>
__global__ __launch_bounds__(256) void gemm_wmma(const _Float16* __restrict__ A,
                                                 const _Float16* __restrict__ W,
                                                 const float* __restrict__ bias,
                                                 const float* __restrict__ resid,
                                                 _Float16* __restrict__ out_h,
                                                 float* __restrict__ out_f,
                                                 int M, int N, int K) {
  __shared__ __align__(16) _Float16 sB[2][64][BPITCH];
  int lane = threadIdx.x & 31;
  int wave = threadIdx.x >> 5;
  int row0 = blockIdx.x * 256 + wave * 32;
  int col0 = blockIdx.y * 64;

  // staging: 64x32 halves, 256 threads x 8 halves (one b128 each)
  int sn = threadIdx.x >> 2;          // 0..63  (column within tile)
  int sk = (threadIdx.x & 3) * 8;     // 0,8,16,24
  const _Float16* wsrc = W + (size_t)(col0 + sn) * K + sk;

#if USE_ASYNC_COPY
  {
    unsigned ldsoff = (unsigned)(uintptr_t)&sB[0][sn][sk];
    asm volatile("global_load_async_to_lds_b128 %0, %1, off"
                 :: "v"(ldsoff), "v"(wsrc) : "memory");
    asm volatile("s_wait_asynccnt 0x0" ::: "memory");
  }
#else
  *(v8h*)&sB[0][sn][sk] = *(const v8h*)wsrc;
#endif
  __syncthreads();

  const _Float16* arow0 = A + (size_t)row0 * K;
  const _Float16* arow1 = A + (size_t)(row0 + 16) * K;
  v8f acc[2][4] = {};
  int nsteps = K / 32;
  for (int t = 0; t < nsteps; ++t) {
    int cur = t & 1;
    bool more = (t + 1) < nsteps;
#if USE_ASYNC_COPY
    if (more) {
      unsigned ldsoff = (unsigned)(uintptr_t)&sB[cur ^ 1][sn][sk];
      const _Float16* g = wsrc + (size_t)(t + 1) * 32;
      asm volatile("global_load_async_to_lds_b128 %0, %1, off"
                   :: "v"(ldsoff), "v"(g) : "memory");
    }
#else
    v8h stage;
    if (more) stage = *(const v8h*)(wsrc + (size_t)(t + 1) * 32);
#endif
    v16h a0 = load_a_frag(arow0, K, t * 32, lane);
    v16h a1 = load_a_frag(arow1, K, t * 32, lane);
#pragma unroll
    for (int j = 0; j < 4; ++j) {
      v16h b = load_b_lds(&sB[cur][0][0], j * 16, lane);
      acc[0][j] = wmma_f16(a0, b, acc[0][j]);
      acc[1][j] = wmma_f16(a1, b, acc[1][j]);
    }
#if USE_ASYNC_COPY
    if (more) asm volatile("s_wait_asynccnt 0x0" ::: "memory");
#else
    if (more) *(v8h*)&sB[cur ^ 1][sn][sk] = stage;
#endif
    __syncthreads();
  }

  int n = lane & 15, hs = lane >> 4;
#pragma unroll
  for (int ms = 0; ms < 2; ++ms) {
#pragma unroll
    for (int j = 0; j < 4; ++j) {
      int col = col0 + j * 16 + n;
      float bv = bias[col];
#pragma unroll
      for (int r = 0; r < 8; ++r) {
        int row = row0 + ms * 16 + r + 8 * hs;
        float v = acc[ms][j][r] + bv;
        if (EPI == 1) v = 0.5f * v * (1.f + erff(v * 0.70710678118654752f));
        if (EPI == 2) {
          size_t idx = (size_t)row * N + col;
          out_f[idx] = v + resid[idx];
        } else {
          out_h[(size_t)row * N + col] = (_Float16)v;
        }
      }
    }
  }
}

// ---------------------------------------------------------------------------
// Edge encoder + scatter-add into edge_bias[H][N][N] (f32, atomics).
// One thread per (edge, head): 512-length dot product.
__global__ __launch_bounds__(256) void edge_scatter_kernel(
    const float* __restrict__ edge_attr, const float* __restrict__ We,
    const float* __restrict__ be, const int* __restrict__ edge_index,
    float* __restrict__ ebias) {
  int idx = blockIdx.x * 256 + threadIdx.x;
  int hh = idx & (H_ - 1);
  int e  = idx >> 3;
  if (e >= NE_) return;
  const float* ea = edge_attr + (size_t)e * E_;
  const float* w  = We + (size_t)hh * E_;
  float acc = 0.f;
  for (int k = 0; k < E_; ++k) acc += ea[k] * w[k];
  int src = edge_index[e];
  int tgt = edge_index[NE_ + e];
  atomicAdd(&ebias[(size_t)hh * N_ * N_ + (size_t)src * N_ + tgt], acc + be[hh]);
}

// ---------------------------------------------------------------------------
// Fused attention: one block (128 thr = 4 waves) per (b, h, 16-query tile).
// Phase 1: S = Q K^T * SCALE + dist + edge  (WMMA, f32 into LDS)
// Phase 2: row softmax, probs -> f16 LDS
// Phase 3: O = P @ V (WMMA, A from LDS), f16 out [B,N,H,DH]
__global__ __launch_bounds__(128) void attn_kernel(
    const _Float16* __restrict__ q16, const _Float16* __restrict__ k16,
    const _Float16* __restrict__ v16, const float* __restrict__ dist_table,
    const int* __restrict__ dist_mat, const float* __restrict__ ebias,
    _Float16* __restrict__ out16) {
  int blk = blockIdx.x;
  int qt = blk & 31;            // N/16 = 32 query tiles
  int bh = blk >> 5;
  int hh = bh & (H_ - 1);
  int b  = bh >> 3;
  int m0 = qt * 16;
  int lane = threadIdx.x & 31;
  int wave = threadIdx.x >> 5;  // 0..3

  __shared__ __align__(16) float    sS[16][N_ + 4];
  __shared__ __align__(16) _Float16 sP[16][N_];

  // --- Phase 1: scores ---
  const _Float16* qrow = q16 + ((size_t)(b * N_ + m0)) * E_ + hh * DH_;
  v16h qa0 = load_a_frag(qrow, E_, 0, lane);
  v16h qa1 = load_a_frag(qrow, E_, 32, lane);
  int nn = lane & 15, hs = lane >> 4;
  for (int jj = 0; jj < 8; ++jj) {
    int n0 = (wave * 8 + jj) * 16;
    const _Float16* kcol = k16 + ((size_t)(b * N_ + n0)) * E_ + hh * DH_;
    v8f acc = {};
    acc = wmma_f16(qa0, load_b_frag(kcol, E_, 0, lane), acc);
    acc = wmma_f16(qa1, load_b_frag(kcol, E_, 32, lane), acc);
#pragma unroll
    for (int r = 0; r < 8; ++r) {
      int mg = m0 + r + 8 * hs;
      int ng = n0 + nn;
      int d = dist_mat[((size_t)b * N_ + mg) * N_ + ng];
      d = d < 0 ? 0 : (d > 99 ? 99 : d);
      float v = acc[r] * SCALE_ + dist_table[d * H_ + hh] +
                ebias[(size_t)hh * N_ * N_ + (size_t)mg * N_ + ng];
      sS[r + 8 * hs][n0 + nn] = v;
    }
  }
  __syncthreads();

  // --- Phase 2: softmax (8 threads per row) ---
  {
    int row = threadIdx.x >> 3;
    int sub = threadIdx.x & 7;
    float mx = -3.4e38f;
    for (int c = sub; c < N_; c += 8) mx = fmaxf(mx, sS[row][c]);
#pragma unroll
    for (int o = 4; o > 0; o >>= 1) mx = fmaxf(mx, __shfl_xor(mx, o, 8));
    float sum = 0.f;
    for (int c = sub; c < N_; c += 8) {
      float e = __expf(sS[row][c] - mx);
      sS[row][c] = e;
      sum += e;
    }
#pragma unroll
    for (int o = 4; o > 0; o >>= 1) sum += __shfl_xor(sum, o, 8);
    float inv = 1.f / sum;
    for (int c = sub; c < N_; c += 8) sP[row][c] = (_Float16)(sS[row][c] * inv);
  }
  __syncthreads();

  // --- Phase 3: P @ V, each wave handles 16 of the 64 head dims ---
  int d0 = wave * 16;
  v8f acc = {};
  for (int k0 = 0; k0 < N_; k0 += 32) {
    v16h a = load_a_frag(&sP[0][0], N_, k0, lane);
    // B[k][j] = V[k0+hs*16+i][d0+n] (strided global gather)
    const _Float16* vb =
        v16 + ((size_t)(b * N_ + k0 + hs * 16)) * E_ + hh * DH_ + d0 + nn;
    v16h bf;
#pragma unroll
    for (int i = 0; i < 16; ++i) bf[i] = vb[(size_t)i * E_];
    acc = wmma_f16(a, bf, acc);
  }
#pragma unroll
  for (int r = 0; r < 8; ++r) {
    int mg = m0 + r + 8 * hs;
    out16[((size_t)(b * N_ + mg)) * E_ + hh * DH_ + d0 + nn] = (_Float16)acc[r];
  }
}

// ---------------------------------------------------------------------------
extern "C" void kernel_launch(void* const* d_in, const int* in_sizes, int n_in,
                              void* d_out, int out_size, void* d_ws, size_t ws_size,
                              hipStream_t stream) {
  (void)in_sizes; (void)n_in; (void)out_size; (void)ws_size;
  const float* x          = (const float*)d_in[0];
  const float* edge_attr  = (const float*)d_in[1];
  const float* Wq = (const float*)d_in[2];  const float* bq = (const float*)d_in[3];
  const float* Wk = (const float*)d_in[4];  const float* bk = (const float*)d_in[5];
  const float* Wv = (const float*)d_in[6];  const float* bv = (const float*)d_in[7];
  const float* Wo = (const float*)d_in[8];  const float* bo = (const float*)d_in[9];
  const float* dist_table = (const float*)d_in[10];
  const float* We = (const float*)d_in[11]; const float* be = (const float*)d_in[12];
  const float* W1 = (const float*)d_in[13]; const float* b1 = (const float*)d_in[14];
  const float* W2 = (const float*)d_in[15]; const float* b2 = (const float*)d_in[16];
  const float* g1 = (const float*)d_in[17]; const float* bt1 = (const float*)d_in[18];
  const float* g2 = (const float*)d_in[19]; const float* bt2 = (const float*)d_in[20];
  const int* edge_index = (const int*)d_in[21];
  const int* dist_mat   = (const int*)d_in[22];

  // workspace carve-up
  char* ws = (char*)d_ws;
  size_t off = 0;
  auto take = [&](size_t bytes) {
    void* p = ws + off;
    off += (bytes + 255) & ~(size_t)255;
    return p;
  };
  _Float16* h16    = (_Float16*)take((size_t)MROWS * E_ * 2);
  _Float16* q16    = (_Float16*)take((size_t)MROWS * E_ * 2);
  _Float16* k16    = (_Float16*)take((size_t)MROWS * E_ * 2);
  _Float16* v16    = (_Float16*)take((size_t)MROWS * E_ * 2);
  _Float16* a16    = (_Float16*)take((size_t)MROWS * E_ * 2);
  _Float16* h2_16  = (_Float16*)take((size_t)MROWS * E_ * 2);
  _Float16* mid16  = (_Float16*)take((size_t)MROWS * 4 * E_ * 2);
  _Float16* wq16   = (_Float16*)take((size_t)E_ * E_ * 2);
  _Float16* wk16   = (_Float16*)take((size_t)E_ * E_ * 2);
  _Float16* wv16   = (_Float16*)take((size_t)E_ * E_ * 2);
  _Float16* wo16   = (_Float16*)take((size_t)E_ * E_ * 2);
  _Float16* w116   = (_Float16*)take((size_t)4 * E_ * E_ * 2);
  _Float16* w216   = (_Float16*)take((size_t)4 * E_ * E_ * 2);
  float*    ebias  = (float*)take((size_t)H_ * N_ * N_ * 4);
  float*    x1     = (float*)take((size_t)MROWS * E_ * 4);

  // 1) weight casts
  int ne = E_ * E_;
  cast_f16_kernel<<<(ne + 255) / 256, 256, 0, stream>>>(Wq, wq16, ne);
  cast_f16_kernel<<<(ne + 255) / 256, 256, 0, stream>>>(Wk, wk16, ne);
  cast_f16_kernel<<<(ne + 255) / 256, 256, 0, stream>>>(Wv, wv16, ne);
  cast_f16_kernel<<<(ne + 255) / 256, 256, 0, stream>>>(Wo, wo16, ne);
  cast_f16_kernel<<<(4 * ne + 255) / 256, 256, 0, stream>>>(W1, w116, 4 * ne);
  cast_f16_kernel<<<(4 * ne + 255) / 256, 256, 0, stream>>>(W2, w216, 4 * ne);

  // 2) zero edge bias
  hipMemsetAsync(ebias, 0, (size_t)H_ * N_ * N_ * 4, stream);

  // 3) LN1
  layernorm_kernel<<<MROWS, 256, 0, stream>>>(x, g1, bt1, h16);

  // 4) Q/K/V projections
  dim3 gE(MROWS / 256, E_ / 64);
  gemm_wmma<0><<<gE, 256, 0, stream>>>(h16, wq16, bq, nullptr, q16, nullptr, MROWS, E_, E_);
  gemm_wmma<0><<<gE, 256, 0, stream>>>(h16, wk16, bk, nullptr, k16, nullptr, MROWS, E_, E_);
  gemm_wmma<0><<<gE, 256, 0, stream>>>(h16, wv16, bv, nullptr, v16, nullptr, MROWS, E_, E_);

  // 5) edge bias scatter
  edge_scatter_kernel<<<(NE_ * H_) / 256, 256, 0, stream>>>(edge_attr, We, be,
                                                            edge_index, ebias);

  // 6) fused attention
  attn_kernel<<<B_ * H_ * (N_ / 16), 128, 0, stream>>>(q16, k16, v16, dist_table,
                                                       dist_mat, ebias, a16);

  // 7) output projection + residual -> x1
  gemm_wmma<2><<<gE, 256, 0, stream>>>(a16, wo16, bo, x, nullptr, x1, MROWS, E_, E_);

  // 8) LN2
  layernorm_kernel<<<MROWS, 256, 0, stream>>>(x1, g2, bt2, h2_16);

  // 9) FFN1 + GELU
  dim3 gF(MROWS / 256, (4 * E_) / 64);
  gemm_wmma<1><<<gF, 256, 0, stream>>>(h2_16, w116, b1, nullptr, mid16, nullptr,
                                       MROWS, 4 * E_, E_);

  // 10) FFN2 + residual -> out
  gemm_wmma<2><<<gE, 256, 0, stream>>>(mid16, w216, b2, x1, nullptr, (float*)d_out,
                                       MROWS, E_, 4 * E_);
}
